// channel_wise_78563541778519
// MI455X (gfx1250) — matrix operand order repaired and checked by reading, not compiled, not run on hardware
//
#include <hip/hip_runtime.h>
#include <hip/hip_bf16.h>

// ---------------------------------------------------------------------------
// channel-wise attention:  S = X X^T ; P = softmax(S) ; Y = gamma * (P X)
// B=16, C=768, HW=4096.  bf16 WMMA (16x16x32) with fp32 accumulate.
// CDNA5 path: global_load_async_to_lds_b128 staging (ASYNCcnt) + double-
// buffered LDS tiles so async copy overlaps v_wmma compute.
// Workspace: [ xh bf16 16*768*4096 | logits f32 16*768*768 | probs bf16 16*768*768 ]
// ---------------------------------------------------------------------------

typedef __attribute__((ext_vector_type(16))) __bf16 v16bf;
typedef __attribute__((ext_vector_type(8)))  __bf16 v8bf;
typedef __attribute__((ext_vector_type(4)))  __bf16 v4bf;
typedef __attribute__((ext_vector_type(8)))  float  v8f;

#define TILE_M 128
#define TILE_N 128
#define TILE_K 32
#define LDA_S  40    // bf16 elems/row: 80 B (16B aligned, 20-bank stride -> conflict free)
#define LDB_S  136   // bf16 elems/row: 272 B (16B aligned, 68-bank stride)

__device__ __forceinline__ unsigned lds_off(const void* p) {
    // generic (flat) shared pointer: low 32 bits are the LDS byte address
    return (unsigned)(uintptr_t)p;
}

__device__ __forceinline__ void async_ld_b128(unsigned lds_addr, const void* gaddr) {
    asm volatile("global_load_async_to_lds_b128 %0, %1, off"
                 :: "v"(lds_addr), "v"(gaddr) : "memory");
}

__device__ __forceinline__ void wait_async0() {
    asm volatile("s_wait_asynccnt 0x0" ::: "memory");
}

// ---------------------------------------------------------------------------
// fp32 -> bf16 conversion (vectorized, grid-stride)
// ---------------------------------------------------------------------------
__global__ __launch_bounds__(256)
void cvt_f32_bf16_kernel(const float4* __restrict__ in, v4bf* __restrict__ out, int n4) {
    int i = blockIdx.x * blockDim.x + threadIdx.x;
    int stride = gridDim.x * blockDim.x;
    for (; i < n4; i += stride) {
        float4 f = in[i];
        v4bf o;
        o[0] = (__bf16)f.x; o[1] = (__bf16)f.y;
        o[2] = (__bf16)f.z; o[3] = (__bf16)f.w;
        out[i] = o;
    }
}

// ---------------------------------------------------------------------------
// batched bf16 GEMM, C(f32) = A(bf16) * B(bf16)[^T], 128x128x32 tiles,
// double-buffered async LDS staging. 8 waves/WG, each wave 32x64
// (2x4 fragments of 16x16), 8 WMMAs per tile per wave.
// TRANS_B=true  : B_logical[k][n] = Bsrc[n*ldb + k]   (Gram: B = X^T)
// TRANS_B=false : B_logical[k][n] = Bsrc[k*ldb + n]
// ---------------------------------------------------------------------------
template<bool TRANS_B, bool APPLY_GAMMA>
__global__ __launch_bounds__(256)
void gemm_bf16_wmma(const __bf16* __restrict__ Aall,
                    const __bf16* __restrict__ Ball,
                    float* __restrict__ Call,
                    const float* __restrict__ gamma,
                    int K, int lda, int ldb, int ldc,
                    long long sA, long long sB, long long sC)
{
    __shared__ __bf16 ldsA[2][TILE_M * LDA_S];   // 2 x 10,240 B
    __shared__ __bf16 ldsB[2][TILE_K * LDB_S];   // 2 x  8,704 B

    const int batch = blockIdx.z;
    const __bf16* A  = Aall + (long long)batch * sA;
    const __bf16* Bm = Ball + (long long)batch * sB;
    float*        Cp = Call + (long long)batch * sC;

    const int tid   = threadIdx.x;
    const int lane  = tid & 31;
    const int wave  = tid >> 5;
    const int waveM = wave >> 1;   // 0..3
    const int waveN = wave & 1;    // 0..1
    const int tileM = blockIdx.y * TILE_M;
    const int tileN = blockIdx.x * TILE_N;

    v8f acc[2][4];
#pragma unroll
    for (int i = 0; i < 2; ++i)
#pragma unroll
        for (int j = 0; j < 4; ++j)
            acc[i][j] = (v8f)0.0f;

    // ---- staging helpers (each thread: 2 x 16B chunks per tile) ----
    auto issueA = [&](int buf, int k0) {
#pragma unroll
        for (int i = 0; i < 2; ++i) {
            int c   = tid + i * 256;          // 0..511
            int row = c >> 2;                 // 0..127
            int col = (c & 3) << 3;           // 0..24
            const void* g = (const void*)(A + (long long)(tileM + row) * lda + k0 + col);
            async_ld_b128(lds_off(&ldsA[buf][row * LDA_S + col]), g);
        }
    };
    auto issueB_nt = [&](int buf, int k0) {
#pragma unroll
        for (int i = 0; i < 2; ++i) {
            int c    = tid + i * 256;
            int krow = c >> 4;                // 0..31
            int col  = (c & 15) << 3;         // 0..120
            const void* g = (const void*)(Bm + (long long)(k0 + krow) * ldb + tileN + col);
            async_ld_b128(lds_off(&ldsB[buf][krow * LDB_S + col]), g);
        }
    };
    // transposed-B path: global->regs (issued early), regs->LDS (after compute)
    v8bf tb[2];
    auto loadB_t = [&](int k0) {
#pragma unroll
        for (int i = 0; i < 2; ++i) {
            int c    = tid + i * 256;
            int nrow = c >> 2;                // 0..127 (N)
            int kcol = (c & 3) << 3;          // 0..24  (K)
            tb[i] = *(const v8bf*)(Bm + (long long)(tileN + nrow) * ldb + k0 + kcol);
        }
    };
    auto storeB_t = [&](int buf) {
#pragma unroll
        for (int i = 0; i < 2; ++i) {
            int c    = tid + i * 256;
            int nrow = c >> 2;
            int kcol = (c & 3) << 3;
#pragma unroll
            for (int j = 0; j < 8; ++j)
                ldsB[buf][(kcol + j) * LDB_S + nrow] = tb[i][j];
        }
    };

    // ---- prologue: stage tile 0 ----
    issueA(0, 0);
    if (TRANS_B) { loadB_t(0); storeB_t(0); }
    else         { issueB_nt(0, 0); }
    wait_async0();
    __syncthreads();

    const int nIter = K / TILE_K;
    for (int it = 0; it < nIter; ++it) {
        const int  buf  = it & 1;
        const int  nxt  = buf ^ 1;
        const bool more = (it + 1) < nIter;
        const int  k0n  = (it + 1) * TILE_K;

        // kick off next tile's copies before computing this one
        if (more) {
            issueA(nxt, k0n);
            if (TRANS_B) loadB_t(k0n);
            else         issueB_nt(nxt, k0n);
        }

        // ---- compute: 2x4 fragments, 8 WMMAs ----
        v16bf afrag[2], bfrag[4];
#pragma unroll
        for (int fm = 0; fm < 2; ++fm) {
            // A 16x32 bf16: lane<16 -> row=lane, K=0..15 ; lane>=16 -> row=lane-16, K=16..31
            const __bf16* p = &ldsA[buf][(waveM * 32 + fm * 16 + (lane & 15)) * LDA_S
                                         + ((lane >> 4) << 4)];
            v8bf lo = *(const v8bf*)p;
            v8bf hi = *(const v8bf*)(p + 8);
            afrag[fm] = __builtin_shufflevector(lo, hi,
                0,1,2,3,4,5,6,7,8,9,10,11,12,13,14,15);
        }
#pragma unroll
        for (int fn = 0; fn < 4; ++fn) {
            // B 32x16 bf16: lane<16 -> col=lane, K=0..15 ; lane>=16 -> col=lane-16, K=16..31
            const __bf16* p = &ldsB[buf][((lane >> 4) << 4) * LDB_S + waveN * 64 + fn * 16 + (lane & 15)];
#pragma unroll
            for (int s = 0; s < 16; ++s)
                bfrag[fn][s] = p[s * LDB_S];
        }
#pragma unroll
        for (int fm = 0; fm < 2; ++fm)
#pragma unroll
            for (int fn = 0; fn < 4; ++fn)
                acc[fm][fn] = __builtin_amdgcn_wmma_f32_16x16x32_bf16(
                    false, afrag[fm], false, bfrag[fn],
                    (short)0, acc[fm][fn], false, false);

        // transposed B: commit next tile to LDS after compute (hides load latency)
        if (more && TRANS_B) storeB_t(nxt);

        wait_async0();       // this wave's async copies into `nxt` have landed
        __syncthreads();     // everyone done reading `buf`, everyone's `nxt` visible
    }

    // ---- epilogue: VGPR r: lanes0-15 M=r, lanes16-31 M=8+r; N=lane&15 ----
    float g = 1.0f;
    if constexpr (APPLY_GAMMA) g = gamma[0];
#pragma unroll
    for (int fm = 0; fm < 2; ++fm)
#pragma unroll
        for (int fn = 0; fn < 4; ++fn) {
            int row0 = tileM + waveM * 32 + fm * 16 + ((lane >> 4) << 3);
            int col  = tileN + waveN * 64 + fn * 16 + (lane & 15);
#pragma unroll
            for (int r = 0; r < 8; ++r) {
                float v = acc[fm][fn][r];
                if constexpr (APPLY_GAMMA) v *= g;
                Cp[(long long)(row0 + r) * ldc + col] = v;
            }
        }
}

// ---------------------------------------------------------------------------
// softmax over rows of length 768 -> bf16 probabilities. One wave per row.
// ---------------------------------------------------------------------------
__global__ __launch_bounds__(256)
void softmax_rows_kernel(const float* __restrict__ S, __bf16* __restrict__ P, int rows) {
    const int lane = threadIdx.x & 31;
    const int wave = threadIdx.x >> 5;
    const int row  = blockIdx.x * 8 + wave;
    if (row >= rows) return;

    const float* src = S + (long long)row * 768;
    float v[24];
    float m = -INFINITY;
#pragma unroll
    for (int i = 0; i < 24; ++i) {
        v[i] = src[lane + i * 32];
        m = fmaxf(m, v[i]);
    }
#pragma unroll
    for (int off = 16; off > 0; off >>= 1)
        m = fmaxf(m, __shfl_xor(m, off, 32));

    float s = 0.0f;
#pragma unroll
    for (int i = 0; i < 24; ++i) {
        v[i] = __expf(v[i] - m);
        s += v[i];
    }
#pragma unroll
    for (int off = 16; off > 0; off >>= 1)
        s += __shfl_xor(s, off, 32);

    float inv = 1.0f / s;
    __bf16* dst = P + (long long)row * 768;
#pragma unroll
    for (int i = 0; i < 24; ++i)
        dst[lane + i * 32] = (__bf16)(v[i] * inv);
}

// ---------------------------------------------------------------------------
extern "C" void kernel_launch(void* const* d_in, const int* in_sizes, int n_in,
                              void* d_out, int out_size, void* d_ws, size_t ws_size,
                              hipStream_t stream) {
    const float* x     = (const float*)d_in[0];   // [16,768,64,64] f32
    const float* gamma = (const float*)d_in[1];   // [1] f32
    float* out = (float*)d_out;                   // [16,768,64,64] f32

    char* ws = (char*)d_ws;
    __bf16* xh     = (__bf16*)ws;                               // 100,663,296 B
    float*  logits = (float*)(ws + 100663296ULL);               //  37,748,736 B
    __bf16* probs  = (__bf16*)(ws + 138412032ULL);              //  18,874,368 B

    const long long sX = 768LL * 4096;   // per-batch X elems
    const long long sS = 768LL * 768;    // per-batch S elems

    // 1) x -> bf16
    const int n4 = (16 * 768 * 4096) / 4;   // 12,582,912
    cvt_f32_bf16_kernel<<<8192, 256, 0, stream>>>((const float4*)x, (v4bf*)xh, n4);

    // 2) S = X X^T  (M=N=768, K=4096), B transposed-from-X
    dim3 g1(768 / TILE_N, 768 / TILE_M, 16);
    gemm_bf16_wmma<true, false><<<g1, 256, 0, stream>>>(
        xh, xh, logits, nullptr,
        /*K=*/4096, /*lda=*/4096, /*ldb=*/4096, /*ldc=*/768,
        sX, sX, sS);

    // 3) P = softmax(S) rows, bf16
    softmax_rows_kernel<<<(16 * 768) / 8, 256, 0, stream>>>(logits, probs, 16 * 768);

    // 4) Y = gamma * (P X)  (M=768, N=4096, K=768), B row-major
    dim3 g2(4096 / TILE_N, 768 / TILE_M, 16);
    gemm_bf16_wmma<false, true><<<g2, 256, 0, stream>>>(
        probs, xh, out, gamma,
        /*K=*/768, /*lda=*/768, /*ldb=*/4096, /*ldc=*/4096,
        sS, sX, sX);
}
